// Dominant_82197084110901
// MI455X (gfx1250) — compile-verified, matrix operand order checked by reading, not run on hardware
//
#include <hip/hip_runtime.h>
#include <math.h>

#define NEG_SLOPE 0.2f
#define BN_EPS    1e-5f

typedef __attribute__((ext_vector_type(16))) _Float16 v16h;
typedef __attribute__((ext_vector_type(8)))  float    v8f;

static inline int cdiv(int a, int b) { return (a + b - 1) / b; }

// ---------------------------------------------------------------------------
// GEMM: H[N x DOUT] = X[N x DIN] * W[DIN x DOUT], fp32 in, f16 WMMA, f32 acc.
// One wave32 computes one 16x16 tile of H via v_wmma_f32_16x16x32_f16.
//
// A-fragment (CDNA5 16-bit A, 16x32): lanes 0-15 hold row M=lane with
//   halves 0..7 -> K=k0+0..7, halves 8..15 -> K=k0+16..23; lanes 16-31 hold
//   the same rows with K offset +8. B mirrors this per output column.
//
// W is staged into LDS *pre-packed in fragment order*: for each
// (k-step, col-tile, lane) the 16 halves of that lane's v16h are contiguous,
// so the inner loop issues 2x ds_load_b128 per fragment instead of
// 16x ds_load_u16 + v_mov_b16 repacking.
// ---------------------------------------------------------------------------
template <int DIN, int DOUT>
__global__ void __launch_bounds__(128)
gat_gemm_wmma(const float* __restrict__ X, const float* __restrict__ W,
              float* __restrict__ H, int n)
{
    constexpr int CT = DOUT / 16;   // column tiles
    constexpr int KS = DIN / 32;    // k-steps
    __shared__ __align__(32) _Float16 sB[KS * CT * 32 * 16];

    // Cooperative pack: idx = ((ks*CT + ct)*32 + lane)*16 + e
    for (int idx = threadIdx.x; idx < KS * CT * 32 * 16; idx += 128) {
        const int e    = idx & 15;
        const int ln   = (idx >> 4) & 31;
        const int ct   = (idx >> 9) % CT;
        const int ks   = idx / (512 * CT);
        const int col  = ct * 16 + (ln & 15);
        const int k    = ks * 32 + (ln >> 4) * 8 + (e & 7) + ((e >> 3) << 4);
        sB[idx] = (_Float16)W[(size_t)k * DOUT + col];
    }
    __syncthreads();

    const int gw      = blockIdx.x * 4 + (threadIdx.x >> 5);
    const int tileRow = gw / CT;
    const int tileCol = gw % CT;
    if (tileRow * 16 >= n) return;

    const int lane = threadIdx.x & 31;
    const int r    = lane & 15;
    const int hi   = lane >> 4;

    const float* xrow  = X + (size_t)(tileRow * 16 + r) * DIN;
    const v16h*  bfrag = (const v16h*)sB + (size_t)tileCol * 32 + lane;

    v8f acc = {};
#pragma unroll
    for (int ks = 0; ks < KS; ++ks) {
        const int k0 = ks * 32;
        v16h a;
#pragma unroll
        for (int e = 0; e < 8; ++e) {
            a[e]     = (_Float16)xrow[k0 + hi * 8 + e];        // global_load_b128 x2
            a[e + 8] = (_Float16)xrow[k0 + 16 + hi * 8 + e];   // + v_cvt_pk_f16_f32
        }
        const v16h b = bfrag[(size_t)ks * CT * 32];            // 2x ds_load_b128
        acc = __builtin_amdgcn_wmma_f32_16x16x32_f16(
            /*neg_a=*/false, a, /*neg_b=*/false, b,
            /*c_mod=*/(short)0, acc, /*reuse_a=*/false, /*reuse_b=*/false);
    }

#pragma unroll
    for (int j = 0; j < 8; ++j) {
        const int m = j + 8 * hi;
        H[(size_t)(tileRow * 16 + m) * DOUT + tileCol * 16 + r] = acc[j];
    }
}

// ---------------------------------------------------------------------------
// Per-node attention scores + softmax-state init.
// ---------------------------------------------------------------------------
template <int DOUT>
__global__ void node_prep(const float* __restrict__ H,
                          const float* __restrict__ a_s,
                          const float* __restrict__ a_d,
                          float* __restrict__ s_src, float* __restrict__ s_dst,
                          float* __restrict__ mmax,  float* __restrict__ denom,
                          int n)
{
    const int i = blockIdx.x * blockDim.x + threadIdx.x;
    if (i >= n) return;
    const float* h = H + (size_t)i * DOUT;
    float ss = 0.f, sd = 0.f;
#pragma unroll 8
    for (int c = 0; c < DOUT; ++c) {
        const float v = h[c];
        ss += v * a_s[c];
        sd += v * a_d[c];
    }
    s_src[i] = ss;
    s_dst[i] = sd;
    mmax[i]  = -INFINITY;
    denom[i] = 0.f;
}

__global__ void zero_f32(float* __restrict__ p, long long total)
{
    const long long i = (long long)blockIdx.x * blockDim.x + threadIdx.x;
    if (i < total) p[i] = 0.f;
}

// Edge pass 1: e = leaky_relu(s_src[s] + s_dst[d]); segment max via f32 atomics.
__global__ void edge_max(const int* __restrict__ ei, int E, int n,
                         const float* __restrict__ s_src,
                         const float* __restrict__ s_dst,
                         float* __restrict__ eb, float* __restrict__ mmax)
{
    const int i  = blockIdx.x * blockDim.x + threadIdx.x;
    const int ET = E + n;
    if (i >= ET) return;
    const int s = (i < E) ? ei[i]     : (i - E);   // self-loops appended
    const int d = (i < E) ? ei[E + i] : (i - E);
    float e = s_src[s] + s_dst[d];
    e = (e > 0.f) ? e : NEG_SLOPE * e;
    eb[i] = e;
    atomicMax(&mmax[d], e);                        // global_atomic_max_num_f32
}

// Edge pass 2: p = exp(e - m[d]); segment sum of p.
__global__ void edge_exp(const int* __restrict__ ei, int E, int n,
                         const float* __restrict__ mmax,
                         float* __restrict__ eb, float* __restrict__ denom)
{
    const int i  = blockIdx.x * blockDim.x + threadIdx.x;
    const int ET = E + n;
    if (i >= ET) return;
    const int d = (i < E) ? ei[E + i] : (i - E);
    const float p = __expf(eb[i] - mmax[d]);
    eb[i] = p;                                     // overwrite e with p
    atomicAdd(&denom[d], p);
}

// Edge pass 3: wave per edge; lanes stride features; Y[d] += alpha * H[s].
// H (25.6-102 MB) is resident in the 192 MB L2, so gathers + f32 atomics
// resolve on-die.
template <int DOUT>
__global__ void edge_agg(const int* __restrict__ ei, int E, int n,
                         const float* __restrict__ eb,
                         const float* __restrict__ denom,
                         const float* __restrict__ H, float* __restrict__ Y)
{
    const int wid  = (blockIdx.x * blockDim.x + threadIdx.x) >> 5;
    const int lane = threadIdx.x & 31;
    const int ET   = E + n;
    if (wid >= ET) return;
    const int s = (wid < E) ? ei[wid]     : (wid - E);
    const int d = (wid < E) ? ei[E + wid] : (wid - E);
    const float* hs = H + (size_t)s * DOUT;
    __builtin_prefetch(hs, 0, 0);                  // global_prefetch_b8
    const float alpha = eb[wid] / (denom[d] + 1e-16f);
    float* yd = Y + (size_t)d * DOUT;
#pragma unroll
    for (int c = lane; c < DOUT; c += 32)
        atomicAdd(&yd[c], alpha * hs[c]);          // global_atomic_add_f32
}

// Epilogue: relu(Y + bias), then eval-mode BatchNorm (layers 1-3).
template <int DOUT, bool BN>
__global__ void epilogue(float* __restrict__ Y, const float* __restrict__ bias,
                         const float* __restrict__ g, const float* __restrict__ be,
                         int n)
{
    const long long idx = (long long)blockIdx.x * blockDim.x + threadIdx.x;
    if (idx >= (long long)n * DOUT) return;
    const int c = (int)(idx & (DOUT - 1));
    float v = Y[idx] + bias[c];
    v = fmaxf(v, 0.f);
    if (BN) {
        const float scale = g[c] * rsqrtf(1.f + BN_EPS);
        v = v * scale + be[c];
    }
    Y[idx] = v;
}

// ---------------------------------------------------------------------------
// One GAT layer.
// ---------------------------------------------------------------------------
template <int DIN, int DOUT, bool BN>
static void run_layer(const float* X, const float* Wm, const float* a_s,
                      const float* a_d, const float* bias, const float* g,
                      const float* be, float* H, float* Y, float* s_src,
                      float* s_dst, float* mmax, float* denom, float* eb,
                      const int* ei, int N, int E, hipStream_t stream)
{
    const int ET = E + N;
    const int totalWaves = cdiv(N, 16) * (DOUT / 16);

    gat_gemm_wmma<DIN, DOUT><<<cdiv(totalWaves, 4), 128, 0, stream>>>(X, Wm, H, N);
    node_prep<DOUT><<<cdiv(N, 256), 256, 0, stream>>>(H, a_s, a_d, s_src, s_dst,
                                                      mmax, denom, N);
    zero_f32<<<cdiv(N * DOUT, 256), 256, 0, stream>>>(Y, (long long)N * DOUT);
    edge_max<<<cdiv(ET, 256), 256, 0, stream>>>(ei, E, N, s_src, s_dst, eb, mmax);
    edge_exp<<<cdiv(ET, 256), 256, 0, stream>>>(ei, E, N, mmax, eb, denom);
    edge_agg<DOUT><<<cdiv(ET * 32, 256), 256, 0, stream>>>(ei, E, N, eb, denom, H, Y);
    epilogue<DOUT, BN><<<cdiv(N * DOUT, 256), 256, 0, stream>>>(Y, bias, g, be, N);
}

extern "C" void kernel_launch(void* const* d_in, const int* in_sizes, int n_in,
                              void* d_out, int out_size, void* d_ws, size_t ws_size,
                              hipStream_t stream)
{
    const int N = in_sizes[0] / 256;       // feat: [N, 256]
    const int E = in_sizes[1] / 2;         // edge_index: [2, E]
    const int ET = E + N;

    const float* feat = (const float*)d_in[0];
    const int*   ei   = (const int*)d_in[1];
    const float* W1  = (const float*)d_in[2];
    const float* as1 = (const float*)d_in[3];
    const float* ad1 = (const float*)d_in[4];
    const float* bi1 = (const float*)d_in[5];
    const float* W2  = (const float*)d_in[6];
    const float* as2 = (const float*)d_in[7];
    const float* ad2 = (const float*)d_in[8];
    const float* bi2 = (const float*)d_in[9];
    const float* W3  = (const float*)d_in[10];
    const float* as3 = (const float*)d_in[11];
    const float* ad3 = (const float*)d_in[12];
    const float* bi3 = (const float*)d_in[13];
    const float* W4  = (const float*)d_in[14];
    const float* as4 = (const float*)d_in[15];
    const float* ad4 = (const float*)d_in[16];
    const float* bi4 = (const float*)d_in[17];
    const float* g1  = (const float*)d_in[18];
    const float* be1 = (const float*)d_in[19];
    const float* g2  = (const float*)d_in[20];
    const float* be2 = (const float*)d_in[21];
    const float* g3  = (const float*)d_in[22];
    const float* be3 = (const float*)d_in[23];
    (void)n_in; (void)out_size; (void)ws_size;

    // Workspace carve-up (floats, 256B-aligned chunks).
    float* ws = (float*)d_ws;
    size_t off = 0;
    auto carve = [&](size_t nf) {
        float* p = ws + off;
        off += (nf + 63) & ~(size_t)63;
        return p;
    };
    float* Hbuf  = carve((size_t)N * 256);   // h for current layer (max width 256)
    float* Y0    = carve((size_t)N * 64);
    float* Y1    = carve((size_t)N * 64);
    float* s_src = carve(N);
    float* s_dst = carve(N);
    float* mmax  = carve(N);
    float* denom = carve(N);
    float* eb    = carve(ET);

    float* out = (float*)d_out;              // [N, 256], layer-4 accumulator

    // Encoder
    run_layer<256, 64, true >(feat, W1, as1, ad1, bi1, g1, be1,
                              Hbuf, Y0, s_src, s_dst, mmax, denom, eb, ei, N, E, stream);
    run_layer< 64, 64, true >(Y0,   W2, as2, ad2, bi2, g2, be2,
                              Hbuf, Y1, s_src, s_dst, mmax, denom, eb, ei, N, E, stream);
    // Decoder
    run_layer< 64, 64, true >(Y1,   W3, as3, ad3, bi3, g3, be3,
                              Hbuf, Y0, s_src, s_dst, mmax, denom, eb, ei, N, E, stream);
    run_layer< 64, 256, false>(Y0,  W4, as4, ad4, bi4, nullptr, nullptr,
                              Hbuf, out, s_src, s_dst, mmax, denom, eb, ei, N, E, stream);
}
